// QuantumAttention_22565758173927
// MI455X (gfx1250) — matrix operand order repaired and checked by reading, not compiled, not run on hardware
//
#include <hip/hip_runtime.h>
#include <hip/hip_bf16.h>

#define S_DIM 2048
#define D_DIM 2048
#define H_NUM 1024
#define B_NUM 2

typedef _Float16 half_t;
typedef _Float16 v16h __attribute__((ext_vector_type(16)));
typedef _Float16 v8h  __attribute__((ext_vector_type(8)));
typedef float    v8f  __attribute__((ext_vector_type(8)));

// ---------------------------------------------------------------------------
// Kernel 1: convert Q -> f16, and build rotated+scaled K' -> f16.
// One thread per (b, head, d): handles the row pair (2h, 2h+1).
// ---------------------------------------------------------------------------
__global__ __launch_bounds__(256)
void qa_prep_qk(const float* __restrict__ Q, const float* __restrict__ K,
                const float* __restrict__ qp,
                half_t* __restrict__ Q16, half_t* __restrict__ K16) {
    const float SCALE = 0.70710678118654752f; // 1/sqrt(2)
    int tid = blockIdx.x * blockDim.x + threadIdx.x;   // B*H*D threads
    int d = tid & (D_DIM - 1);
    int rest = tid >> 11;           // /D_DIM
    int h = rest & (H_NUM - 1);
    int b = rest >> 10;             // /H_NUM
    if (b >= B_NUM) return;

    float ang = qp[h * 4];                       // first param column
    float sig = 1.0f / (1.0f + __expf(-ang));    // sigmoid
    float c  = __cosf(sig) * SCALE;
    float sn = __sinf(sig) * SCALE;

    size_t r0 = ((size_t)b * S_DIM + 2 * h) * D_DIM + d;
    size_t r1 = r0 + D_DIM;
    float k0 = K[r0], k1 = K[r1];
    K16[r0] = (half_t)(c * k0 + sn * k1);   // K'_{2h}
    K16[r1] = (half_t)(c * k1 - sn * k0);   // K'_{2h+1}
    Q16[r0] = (half_t)Q[r0];
    Q16[r1] = (half_t)Q[r1];
}

// ---------------------------------------------------------------------------
// Kernel 2: LDS-tiled transpose V (B,S,D) f32 -> Vt (B,D,S) f16.
// ---------------------------------------------------------------------------
__global__ __launch_bounds__(256)
void qa_vtrans(const float* __restrict__ V, half_t* __restrict__ Vt) {
    __shared__ float tile[16][17];
    int b  = blockIdx.z;
    int t0 = blockIdx.x * 16;
    int d0 = blockIdx.y * 16;
    int tx = threadIdx.x, ty = threadIdx.y;
    tile[ty][tx] = V[((size_t)b * S_DIM + t0 + ty) * D_DIM + d0 + tx];
    __syncthreads();
    Vt[((size_t)b * D_DIM + d0 + ty) * S_DIM + t0 + tx] = (half_t)tile[tx][ty];
}

// ---------------------------------------------------------------------------
// Kernel 3: fused S = Q16 * K16^T (f16 WMMA, f32 acc) + row softmax -> P f16.
// Grid: B*(S/32) workgroups of 512 threads (16 waves).
// Wave w: rows [m0, m0+32) (2 M-tiles), cols [w*128, w*128+128) (8 N-tiles).
// Each B fragment feeds 2 WMMAs -> 20 b128 loads per 16 WMMAs.
// ---------------------------------------------------------------------------
__global__ __launch_bounds__(512)
void qa_qk_softmax(const half_t* __restrict__ Q16,
                   const half_t* __restrict__ K16,
                   half_t* __restrict__ P16) {
    int b    = blockIdx.x / (S_DIM / 32);
    int m0   = (blockIdx.x % (S_DIM / 32)) * 32;
    int wave = threadIdx.x >> 5;           // 0..15
    int lane = threadIdx.x & 31;
    int lrow = lane & 15;
    int hi   = lane >> 4;
    int nbase = wave * 128;

    const half_t* qrow0 = Q16 + ((size_t)b * S_DIM + m0 + lrow) * D_DIM + 16 * hi;
    const half_t* qrow1 = qrow0 + (size_t)16 * D_DIM;
    const half_t* kwave = K16 + (size_t)b * S_DIM * D_DIM
                              + (size_t)(nbase + lrow) * D_DIM + 16 * hi;

    v8f acc[2][8];
#pragma unroll
    for (int mt = 0; mt < 2; ++mt)
#pragma unroll
        for (int nt = 0; nt < 8; ++nt) acc[mt][nt] = (v8f){0,0,0,0,0,0,0,0};

#pragma unroll 1
    for (int d0 = 0; d0 < D_DIM; d0 += 32) {
        // Two A fragments (rows m0+lrow and m0+16+lrow)
        v8h a0lo = *(const v8h*)(qrow0 + d0);
        v8h a0hi = *(const v8h*)(qrow0 + d0 + 8);
        v8h a1lo = *(const v8h*)(qrow1 + d0);
        v8h a1hi = *(const v8h*)(qrow1 + d0 + 8);
        v16h a0 = __builtin_shufflevector(a0lo, a0hi, 0,1,2,3,4,5,6,7,
                                                      8,9,10,11,12,13,14,15);
        v16h a1 = __builtin_shufflevector(a1lo, a1hi, 0,1,2,3,4,5,6,7,
                                                      8,9,10,11,12,13,14,15);
        __builtin_prefetch(qrow0 + d0 + 64, 0, 0);
        __builtin_prefetch(qrow1 + d0 + 64, 0, 0);
#pragma unroll
        for (int nt = 0; nt < 8; ++nt) {
            v16h bf = *(const v16h*)(kwave + (size_t)nt * 16 * D_DIM + d0);
            acc[0][nt] = __builtin_amdgcn_wmma_f32_16x16x32_f16(
                false, a0, false, bf, (short)0, acc[0][nt], false, false);
            acc[1][nt] = __builtin_amdgcn_wmma_f32_16x16x32_f16(
                false, a1, false, bf, (short)0, acc[1][nt], false, false);
        }
    }

    // ---- row softmax over 2048 columns (8 tiles here x 16 waves) ----
    __shared__ float redmax[32][16];
    __shared__ float redsum[32][16];

    float pm[2][8];
#pragma unroll
    for (int mt = 0; mt < 2; ++mt) {
#pragma unroll
        for (int r = 0; r < 8; ++r) {
            float m = acc[mt][0][r];
#pragma unroll
            for (int nt = 1; nt < 8; ++nt) m = fmaxf(m, acc[mt][nt][r]);
#pragma unroll
            for (int mask = 1; mask < 16; mask <<= 1)
                m = fmaxf(m, __shfl_xor(m, mask, 32));
            pm[mt][r] = m;
        }
    }
    if (lrow == 0) {
#pragma unroll
        for (int mt = 0; mt < 2; ++mt)
#pragma unroll
            for (int r = 0; r < 8; ++r)
                redmax[mt * 16 + 8 * hi + r][wave] = pm[mt][r];
    }
    __syncthreads();

    float gm[2][8], ps[2][8];
#pragma unroll
    for (int mt = 0; mt < 2; ++mt) {
#pragma unroll
        for (int r = 0; r < 8; ++r) {
            float m = redmax[mt * 16 + 8 * hi + r][0];
#pragma unroll
            for (int w = 1; w < 16; ++w)
                m = fmaxf(m, redmax[mt * 16 + 8 * hi + r][w]);
            gm[mt][r] = m;
            ps[mt][r] = 0.0f;
        }
    }
#pragma unroll
    for (int mt = 0; mt < 2; ++mt)
#pragma unroll
        for (int nt = 0; nt < 8; ++nt)
#pragma unroll
            for (int r = 0; r < 8; ++r) {
                float e = __expf(acc[mt][nt][r] - gm[mt][r]);
                acc[mt][nt][r] = e;
                ps[mt][r] += e;
            }
#pragma unroll
    for (int mt = 0; mt < 2; ++mt)
#pragma unroll
        for (int r = 0; r < 8; ++r)
#pragma unroll
            for (int mask = 1; mask < 16; mask <<= 1)
                ps[mt][r] += __shfl_xor(ps[mt][r], mask, 32);
    if (lrow == 0) {
#pragma unroll
        for (int mt = 0; mt < 2; ++mt)
#pragma unroll
            for (int r = 0; r < 8; ++r)
                redsum[mt * 16 + 8 * hi + r][wave] = ps[mt][r];
    }
    __syncthreads();

    float inv[2][8];
#pragma unroll
    for (int mt = 0; mt < 2; ++mt) {
#pragma unroll
        for (int r = 0; r < 8; ++r) {
            float s = 0.0f;
#pragma unroll
            for (int w = 0; w < 16; ++w)
                s += redsum[mt * 16 + 8 * hi + r][w];
            inv[mt][r] = 1.0f / s;
        }
    }

    // Write P (aliases Q16 region: same 32 rows, fully read above this point)
    half_t* Prow = P16 + ((size_t)b * S_DIM + m0) * S_DIM;
#pragma unroll
    for (int mt = 0; mt < 2; ++mt)
#pragma unroll
        for (int nt = 0; nt < 8; ++nt)
#pragma unroll
            for (int r = 0; r < 8; ++r)
                Prow[(size_t)(mt * 16 + 8 * hi + r) * S_DIM
                     + nbase + nt * 16 + lrow] =
                    (half_t)(acc[mt][nt][r] * inv[mt][r]);
}

// ---------------------------------------------------------------------------
// Kernel 4: O = P (f16) * V stored as (b,d,t) f16 -> f32 out.
// Same 32x128-per-wave WMMA blocking; K-dim is t (2048).
// ---------------------------------------------------------------------------
__global__ __launch_bounds__(512)
void qa_pv(const half_t* __restrict__ P16,
           const half_t* __restrict__ Vt16,
           float* __restrict__ Out) {
    int b    = blockIdx.x / (S_DIM / 32);
    int m0   = (blockIdx.x % (S_DIM / 32)) * 32;
    int wave = threadIdx.x >> 5;
    int lane = threadIdx.x & 31;
    int lrow = lane & 15;
    int hi   = lane >> 4;
    int nbase = wave * 128;

    const half_t* prow0 = P16 + ((size_t)b * S_DIM + m0 + lrow) * S_DIM + 16 * hi;
    const half_t* prow1 = prow0 + (size_t)16 * S_DIM;
    const half_t* vwave = Vt16 + (size_t)b * D_DIM * S_DIM
                               + (size_t)(nbase + lrow) * S_DIM + 16 * hi;

    v8f acc[2][8];
#pragma unroll
    for (int mt = 0; mt < 2; ++mt)
#pragma unroll
        for (int nt = 0; nt < 8; ++nt) acc[mt][nt] = (v8f){0,0,0,0,0,0,0,0};

#pragma unroll 1
    for (int t0 = 0; t0 < S_DIM; t0 += 32) {
        v8h a0lo = *(const v8h*)(prow0 + t0);
        v8h a0hi = *(const v8h*)(prow0 + t0 + 8);
        v8h a1lo = *(const v8h*)(prow1 + t0);
        v8h a1hi = *(const v8h*)(prow1 + t0 + 8);
        v16h a0 = __builtin_shufflevector(a0lo, a0hi, 0,1,2,3,4,5,6,7,
                                                      8,9,10,11,12,13,14,15);
        v16h a1 = __builtin_shufflevector(a1lo, a1hi, 0,1,2,3,4,5,6,7,
                                                      8,9,10,11,12,13,14,15);
        __builtin_prefetch(prow0 + t0 + 64, 0, 0);
        __builtin_prefetch(prow1 + t0 + 64, 0, 0);
#pragma unroll
        for (int nt = 0; nt < 8; ++nt) {
            v16h bf = *(const v16h*)(vwave + (size_t)nt * 16 * S_DIM + t0);
            acc[0][nt] = __builtin_amdgcn_wmma_f32_16x16x32_f16(
                false, a0, false, bf, (short)0, acc[0][nt], false, false);
            acc[1][nt] = __builtin_amdgcn_wmma_f32_16x16x32_f16(
                false, a1, false, bf, (short)0, acc[1][nt], false, false);
        }
    }

    float* orow = Out + ((size_t)b * S_DIM + m0) * D_DIM;
#pragma unroll
    for (int mt = 0; mt < 2; ++mt)
#pragma unroll
        for (int nt = 0; nt < 8; ++nt)
#pragma unroll
            for (int r = 0; r < 8; ++r)
                orow[(size_t)(mt * 16 + 8 * hi + r) * D_DIM
                     + nbase + nt * 16 + lrow] = acc[mt][nt][r];
}

// ---------------------------------------------------------------------------
extern "C" void kernel_launch(void* const* d_in, const int* in_sizes, int n_in,
                              void* d_out, int out_size, void* d_ws, size_t ws_size,
                              hipStream_t stream) {
    (void)in_sizes; (void)n_in; (void)out_size; (void)ws_size;

    const float* Q  = (const float*)d_in[0];
    const float* K  = (const float*)d_in[1];
    const float* V  = (const float*)d_in[2];
    const float* qp = (const float*)d_in[3];

    char* base = (char*)d_ws;
    const size_t MB16 = (size_t)B_NUM * S_DIM * D_DIM * sizeof(half_t); // 16 MB
    half_t* Q16  = (half_t*)(base);            // later reused as P16
    half_t* K16  = (half_t*)(base + MB16);
    half_t* Vt16 = (half_t*)(base + 2 * MB16);
    half_t* P16  = Q16;                        // alias (safe: per-row ownership)

    // 1) rotate+scale K, convert Q
    int n_prep = B_NUM * H_NUM * D_DIM;        // 4,194,304
    qa_prep_qk<<<n_prep / 256, 256, 0, stream>>>(Q, K, qp, Q16, K16);

    // 2) transpose V to (b, d, t) f16
    qa_vtrans<<<dim3(S_DIM / 16, D_DIM / 16, B_NUM), dim3(16, 16), 0, stream>>>(
        V, Vt16);

    // 3) fused QK'^T + softmax -> P
    qa_qk_softmax<<<B_NUM * (S_DIM / 32), 512, 0, stream>>>(Q16, K16, P16);

    // 4) O = P * V
    qa_pv<<<B_NUM * (S_DIM / 32), 512, 0, stream>>>(P16, Vt16, (float*)d_out);
}
